// SparseConvNet_858993459328
// MI455X (gfx1250) — matrix operand order repaired and checked by reading, not compiled, not run on hardware
//
#include <hip/hip_runtime.h>
#include <hip/hip_bf16.h>

typedef __bf16 bf16_t;
typedef __attribute__((ext_vector_type(16))) __bf16 v16bf;
typedef __attribute__((ext_vector_type(2)))  __bf16 v2bf;
typedef __attribute__((ext_vector_type(8)))  float  v8f;
typedef __attribute__((ext_vector_type(8)))  unsigned int v8u;
typedef __attribute__((ext_vector_type(4)))  int v4i;

#define AS1 __attribute__((address_space(1)))
#define AS3 __attribute__((address_space(3)))
typedef AS1 v4i* g_v4i_ptr;
typedef AS3 v4i* l_v4i_ptr;

#if defined(__AMDGCN__) && __has_builtin(__builtin_amdgcn_global_load_async_to_lds_b128)
#define USE_ASYNC_LDS 1
#else
#define USE_ASYNC_LDS 0
#endif

__device__ __forceinline__ void wait_async_all() {
#if USE_ASYNC_LDS
#if __has_builtin(__builtin_amdgcn_s_wait_asynccnt)
  __builtin_amdgcn_s_wait_asynccnt(0);
#else
  asm volatile("s_wait_asynccnt 0x0" ::: "memory");
#endif
#endif
}

// ---------------------------------------------------------------------------
// Init: x (f32, [C][D][H][W]) * mask -> feat bf16 channels-last [D][H][W][C]
// ---------------------------------------------------------------------------
__global__ void init_feat_kernel(const float* __restrict__ x,
                                 const unsigned char* __restrict__ m,
                                 bf16_t* __restrict__ feat, int D, int C) {
  int vox = blockIdx.x * blockDim.x + threadIdx.x;
  int nvox = D * D * D;
  if (vox >= nvox) return;
  float mm = m[vox] ? 1.0f : 0.0f;
  for (int c = 0; c < C; ++c) {
    feat[vox * C + c] = (bf16_t)(x[c * nvox + vox] * mm);
  }
}

// ---------------------------------------------------------------------------
// Mask dilate: max-pool 3x3x3 stride 2 pad 1 (uint8)
// ---------------------------------------------------------------------------
__global__ void dilate_kernel(const unsigned char* __restrict__ in,
                              unsigned char* __restrict__ out,
                              int Din, int Dout) {
  int o = blockIdx.x * blockDim.x + threadIdx.x;
  int nvox = Dout * Dout * Dout;
  if (o >= nvox) return;
  int z = o / (Dout * Dout);
  int r = o - z * Dout * Dout;
  int y = r / Dout;
  int x = r - y * Dout;
  unsigned char v = 0;
  for (int dz = -1; dz <= 1; ++dz)
    for (int dy = -1; dy <= 1; ++dy)
      for (int dx = -1; dx <= 1; ++dx) {
        int iz = 2 * z + dz, iy = 2 * y + dy, ix = 2 * x + dx;
        if (iz >= 0 && iz < Din && iy >= 0 && iy < Din && ix >= 0 && ix < Din)
          if (in[(iz * Din + iy) * Din + ix]) v = 1;
      }
  out[o] = v;
}

// ---------------------------------------------------------------------------
// Weight transpose + f32->bf16: w [Cout][Cin][27] -> wT [27][Cout][Cin]
// ---------------------------------------------------------------------------
__global__ void wtrans_kernel(const float* __restrict__ w,
                              bf16_t* __restrict__ wT, int Cout, int Cin) {
  int i = blockIdx.x * blockDim.x + threadIdx.x;
  int tot = Cout * Cin * 27;
  if (i >= tot) return;
  int tap = i % 27;
  int r = i / 27;
  int ci = r % Cin;
  int co = r / Cin;
  wT[(tap * Cout + co) * Cin + ci] = (bf16_t)w[i];
}

// ---------------------------------------------------------------------------
// Implicit-GEMM 3D conv, WMMA bf16 -> f32, fused BN+ReLU+mask, bf16 out.
// Block: 128 threads = 4 waves; each wave: 16 output voxels x COUT channels.
// Per-tap weight panel (COUT x CIN bf16) double-buffered in LDS; tap t+1 is
// staged (async-to-LDS when available) before computing tap t.
// ---------------------------------------------------------------------------
template <int COUT, int CIN>
__global__ __launch_bounds__(128)
void conv3d_wmma_kernel(const bf16_t* __restrict__ fin,
                        const bf16_t* __restrict__ wT,   // [27][COUT][CIN]
                        const float* __restrict__ scale,
                        const float* __restrict__ bias,
                        const unsigned char* __restrict__ maskOut,
                        bf16_t* __restrict__ fout,
                        int Din, int Dout, int stride) {
  constexpr int NT   = COUT / 16;     // N tiles of 16
  constexpr int CINH = CIN / 2;       // row stride in dwords
  constexpr int T    = COUT * CINH;   // panel size in dwords
  constexpr int ITER = T / (128 * 4); // b128 granules per thread

  const int tid  = threadIdx.x;
  const int lane = tid & 31;
  const int wave = tid >> 5;
  const int hf   = lane >> 4;
  const int l15  = lane & 15;

  const int tileBase = blockIdx.x * 64 + wave * 16;

  // A-row voxel for this lane (M = l15)
  int mvox = tileBase + l15;
  int az = mvox / (Dout * Dout);
  int rem = mvox - az * Dout * Dout;
  int ay = rem / Dout;
  int ax = rem - ay * Dout;

  v8f acc[NT];
#pragma unroll
  for (int t = 0; t < NT; ++t) acc[t] = (v8f){0, 0, 0, 0, 0, 0, 0, 0};

  __shared__ unsigned int ldsB[2][T];  // double-buffered tap panel

  const unsigned int* fin32 = (const unsigned int*)fin;
  const unsigned int* wT32  = (const unsigned int*)wT;

  auto stage = [&](int buf, int tap) {
    const unsigned int* src = wT32 + tap * T;
#if USE_ASYNC_LDS
#pragma unroll
    for (int it = 0; it < ITER; ++it) {
      int idx = (tid + it * 128) * 4;
      __builtin_amdgcn_global_load_async_to_lds_b128(
          (g_v4i_ptr)(src + idx), (l_v4i_ptr)&ldsB[buf][idx], 0, 0);
    }
#else
#pragma unroll
    for (int it = 0; it < ITER; ++it) {
      int idx = (tid + it * 128) * 4;
      *(uint4*)&ldsB[buf][idx] = *(const uint4*)(src + idx);
    }
#endif
  };

  stage(0, 0);
  wait_async_all();
  __syncthreads();

  for (int tap = 0; tap < 27; ++tap) {
    const int buf = tap & 1;
    if (tap + 1 < 27) stage(buf ^ 1, tap + 1);

    int dz = tap / 9, dy = (tap / 3) % 3, dx = tap % 3;
    int iz = az * stride + dz - 1;
    int iy = ay * stride + dy - 1;
    int ix = ax * stride + dx - 1;
    bool vvalid = (iz >= 0) & (iz < Din) & (iy >= 0) & (iy < Din) &
                  (ix >= 0) & (ix < Din);
    int baseIn = vvalid ? ((iz * Din + iy) * Din + ix) * CINH : 0;

#pragma unroll
    for (int cb = 0; cb < CIN; cb += 32) {
      // --- A fragment: two b128 loads under one validity branch ---
      uint4 a0 = {0, 0, 0, 0}, a1 = {0, 0, 0, 0};
      if (vvalid) {
        int ab = baseIn + (cb >> 1) + 4 * hf;
        a0 = *(const uint4*)&fin32[ab];
        a1 = *(const uint4*)&fin32[ab + 8];
      }
      v8u au;
      au[0] = a0.x; au[1] = a0.y; au[2] = a0.z; au[3] = a0.w;
      au[4] = a1.x; au[5] = a1.y; au[6] = a1.z; au[7] = a1.w;
      v16bf afrag = __builtin_bit_cast(v16bf, au);

      // --- B fragments from LDS (two b128 reads) + WMMA ---
#pragma unroll
      for (int t = 0; t < NT; ++t) {
        int n = l15 + t * 16;
        int bb = n * CINH + (cb >> 1) + 4 * hf;
        uint4 b0 = *(const uint4*)&ldsB[buf][bb];
        uint4 b1 = *(const uint4*)&ldsB[buf][bb + 8];
        v8u bu;
        bu[0] = b0.x; bu[1] = b0.y; bu[2] = b0.z; bu[3] = b0.w;
        bu[4] = b1.x; bu[5] = b1.y; bu[6] = b1.z; bu[7] = b1.w;
        v16bf bfrag = __builtin_bit_cast(v16bf, bu);
        acc[t] = __builtin_amdgcn_wmma_f32_16x16x32_bf16(
            false, afrag, false, bfrag, (short)0, acc[t], false, false);
      }
    }

    wait_async_all();     // next panel resident before tap+1
    __syncthreads();      // and all waves done reading this panel
  }

  // Epilogue: D layout -> lane: N=l15(+16t), VGPR j: M = j + 8*hf
#pragma unroll
  for (int t = 0; t < NT; ++t) {
    int n = l15 + t * 16;
    float s = scale[n];
    float b = bias[n];
#pragma unroll
    for (int j = 0; j < 8; ++j) {
      int m = j + 8 * hf;
      int vox = tileBase + m;
      float v = acc[t][j] * s + b;
      v = v > 0.0f ? v : 0.0f;
      v *= maskOut[vox] ? 1.0f : 0.0f;
      fout[vox * COUT + n] = (bf16_t)v;
    }
  }
}

// ---------------------------------------------------------------------------
// Trilinear sampling: feat bf16 [D][D][D][C] at N points -> out f32 [384][N]
// ---------------------------------------------------------------------------
__global__ void sample_kernel(const bf16_t* __restrict__ feat,
                              const float* __restrict__ grid,
                              float* __restrict__ out,
                              int D, int C, int c0, int N) {
  int i = blockIdx.x * blockDim.x + threadIdx.x;
  if (i >= N) return;
  float fx = (grid[i * 3 + 0] + 1.0f) * 0.5f * (float)(D - 1);
  float fy = (grid[i * 3 + 1] + 1.0f) * 0.5f * (float)(D - 1);
  float fz = (grid[i * 3 + 2] + 1.0f) * 0.5f * (float)(D - 1);
  float x0 = floorf(fx), y0 = floorf(fy), z0 = floorf(fz);

  int   off[8];
  float wgt[8];
  int k = 0;
  for (int dz = 0; dz <= 1; ++dz)
    for (int dy = 0; dy <= 1; ++dy)
      for (int dx = 0; dx <= 1; ++dx) {
        float xi = x0 + dx, yi = y0 + dy, zi = z0 + dz;
        float w = (1.0f - fabsf(fx - xi)) * (1.0f - fabsf(fy - yi)) *
                  (1.0f - fabsf(fz - zi));
        bool valid = (xi >= 0.0f) & (xi < (float)D) & (yi >= 0.0f) &
                     (yi < (float)D) & (zi >= 0.0f) & (zi < (float)D);
        int xc = min(max((int)xi, 0), D - 1);
        int yc = min(max((int)yi, 0), D - 1);
        int zc = min(max((int)zi, 0), D - 1);
        off[k] = ((zc * D + yc) * D + xc) * C;
        wgt[k] = valid ? w : 0.0f;
        ++k;
      }

  const unsigned int* f32p = (const unsigned int*)feat;
  for (int c = 0; c < C; c += 2) {
    float s0 = 0.0f, s1 = 0.0f;
#pragma unroll
    for (int q = 0; q < 8; ++q) {
      unsigned int u = f32p[(off[q] >> 1) + (c >> 1)];
      v2bf p = __builtin_bit_cast(v2bf, u);
      s0 += wgt[q] * (float)p[0];
      s1 += wgt[q] * (float)p[1];
    }
    out[(c0 + c) * N + i]     = s0;
    out[(c0 + c + 1) * N + i] = s1;
  }
}

// ---------------------------------------------------------------------------
extern "C" void kernel_launch(void* const* d_in, const int* in_sizes, int n_in,
                              void* d_out, int out_size, void* d_ws, size_t ws_size,
                              hipStream_t stream) {
  (void)in_sizes; (void)n_in; (void)out_size; (void)ws_size;

  const float*         x    = (const float*)d_in[0];
  const float*         gco  = (const float*)d_in[1];
  const unsigned char* m0   = (const unsigned char*)d_in[2];
  float*               out  = (float*)d_out;
  auto P = [&](int i) { return (const float*)d_in[i]; };

  // workspace carve-up
  char* ws = (char*)d_ws;
  bf16_t*        featA = (bf16_t*)(ws + 0);            // 32 MiB
  bf16_t*        featB = (bf16_t*)(ws + 33554432);     // 32 MiB
  bf16_t*        wT    = (bf16_t*)(ws + 67108864);     // <= 884,736 B
  unsigned char* m1    = (unsigned char*)(ws + 68157440);  // 32768
  unsigned char* m2    = m1 + 32768 + 256;
  unsigned char* m3    = m2 + 4096 + 256;
  unsigned char* m4    = m3 + 512 + 256;

  const int N = 16384;

  // 1) masked input -> bf16 channels-last
  init_feat_kernel<<<dim3((262144 + 255) / 256), dim3(256), 0, stream>>>(
      x, m0, featA, 64, 64);

  // 2) mask pyramid
  dilate_kernel<<<dim3((32768 + 255) / 256), dim3(256), 0, stream>>>(m0, m1, 64, 32);
  dilate_kernel<<<dim3((4096 + 255) / 256),  dim3(256), 0, stream>>>(m1, m2, 32, 16);
  dilate_kernel<<<dim3((512 + 255) / 256),   dim3(256), 0, stream>>>(m2, m3, 16, 8);
  dilate_kernel<<<dim3(1),                   dim3(256), 0, stream>>>(m3, m4, 8, 4);

  auto run_conv = [&](int wi, const bf16_t* fin, bf16_t* fout,
                      const unsigned char* mo, int Din, int Dout,
                      int Cin, int Cout, int stride) {
    int tot = Cout * Cin * 27;
    wtrans_kernel<<<dim3((tot + 255) / 256), dim3(256), 0, stream>>>(
        P(wi), wT, Cout, Cin);
    int blocks = (Dout * Dout * Dout) / 64;
    if (Cout == 64 && Cin == 64)
      conv3d_wmma_kernel<64, 64><<<dim3(blocks), dim3(128), 0, stream>>>(
          fin, wT, P(wi + 1), P(wi + 2), mo, fout, Din, Dout, stride);
    else if (Cout == 128 && Cin == 64)
      conv3d_wmma_kernel<128, 64><<<dim3(blocks), dim3(128), 0, stream>>>(
          fin, wT, P(wi + 1), P(wi + 2), mo, fout, Din, Dout, stride);
    else
      conv3d_wmma_kernel<128, 128><<<dim3(blocks), dim3(128), 0, stream>>>(
          fin, wT, P(wi + 1), P(wi + 2), mo, fout, Din, Dout, stride);
  };
  auto run_sample = [&](const bf16_t* f, int D, int C, int c0) {
    sample_kernel<<<dim3((N + 255) / 256), dim3(256), 0, stream>>>(
        f, gco, out, D, C, c0, N);
  };

  // params (pytree sorted-key order): conv0..conv4 then down0..down3
  // conv0: 3..8  conv1: 9..14  conv2: 15..23  conv3: 24..32  conv4: 33..41
  // down0: 42   down1: 45   down2: 48   down3: 51

  run_conv(3,  featA, featB, m0, 64, 64, 64, 64, 1);   // conv0.0
  run_conv(6,  featB, featA, m0, 64, 64, 64, 64, 1);   // conv0.1
  run_conv(42, featA, featB, m1, 64, 32, 64, 64, 2);   // down0
  run_conv(9,  featB, featA, m1, 32, 32, 64, 64, 1);   // conv1.0
  run_conv(12, featA, featB, m1, 32, 32, 64, 64, 1);   // conv1.1
  run_sample(featB, 32, 64, 0);                        // f1
  run_conv(45, featB, featA, m2, 32, 16, 64, 64, 2);   // down1
  run_conv(15, featA, featB, m2, 16, 16, 64, 64, 1);   // conv2.0
  run_conv(18, featB, featA, m2, 16, 16, 64, 64, 1);   // conv2.1
  run_conv(21, featA, featB, m2, 16, 16, 64, 64, 1);   // conv2.2
  run_sample(featB, 16, 64, 64);                       // f2
  run_conv(48, featB, featA, m3, 16, 8, 64, 128, 2);   // down2
  run_conv(24, featA, featB, m3, 8, 8, 128, 128, 1);   // conv3.0
  run_conv(27, featB, featA, m3, 8, 8, 128, 128, 1);   // conv3.1
  run_conv(30, featA, featB, m3, 8, 8, 128, 128, 1);   // conv3.2
  run_sample(featB, 8, 128, 128);                      // f3
  run_conv(51, featB, featA, m4, 8, 4, 128, 128, 2);   // down3
  run_conv(33, featA, featB, m4, 4, 4, 128, 128, 1);   // conv4.0
  run_conv(36, featB, featA, m4, 4, 4, 128, 128, 1);   // conv4.1
  run_conv(39, featA, featB, m4, 4, 4, 128, 128, 1);   // conv4.2
  run_sample(featB, 4, 128, 256);                      // f4
}
